// NTXentCrossEntropyLoss_32736240730749
// MI455X (gfx1250) — matrix operand order repaired and checked by reading, not compiled, not run on hardware
//
#include <hip/hip_runtime.h>
#include <hip/hip_bf16.h>

// ---------------------------------------------------------------------------
// NT-Xent loss, fused: never materialize the 8192x8192 similarity matrix.
//   K1: interleave + L2-normalize rows; emit bf16 hi/lo split + selfdot
//   K2: fp32 positive-pair dots
//   K3: Z*Z^T via v_wmma_f32_16x16x32_bf16 (3-term hi/lo compensation),
//       fused exp(2x) + row-sum epilogue, deterministic column partials
//   K4: final loss reduction
// ---------------------------------------------------------------------------

typedef __attribute__((ext_vector_type(16))) __bf16       v16bf;
typedef __attribute__((ext_vector_type(8)))  float        v8f;
typedef __attribute__((ext_vector_type(4)))  unsigned int v4u;

union Frag {
    v16bf bf;
    v4u   q[2];   // 2 x 16B = 8 VGPRs = 16 bf16 per lane
};

#define NROWS 8192
#define DIM   256
#define CSPLIT 8          // column partials per row (deterministic, no atomics)

// --------------------------- K1: normalize + bf16 split --------------------
__global__ __launch_bounds__(256)
void ntx_normalize_kernel(const float* __restrict__ z1,
                          const float* __restrict__ z2,
                          __bf16* __restrict__ Zh,
                          __bf16* __restrict__ Zl,
                          float* __restrict__ selfdot) {
    const int row = blockIdx.x;          // 0..8191  (even->z1, odd->z2)
    const int t   = threadIdx.x;         // 0..255 = one element of the row
    const float* src = (row & 1) ? z2 : z1;
    const float x = src[(row >> 1) * DIM + t];

    __shared__ float red[256];
    __shared__ float nrm;
    red[t] = x * x;
    __syncthreads();
    #pragma unroll
    for (int s = 128; s > 0; s >>= 1) {
        if (t < s) red[t] += red[t + s];
        __syncthreads();
    }
    if (t == 0) nrm = fmaxf(sqrtf(red[0]), 1e-12f);
    __syncthreads();

    const float r = x / nrm;
    const __bf16 h = (__bf16)r;                 // RNE high part
    const __bf16 l = (__bf16)(r - (float)h);    // residual low part
    Zh[row * DIM + t] = h;
    Zl[row * DIM + t] = l;

    // diagonal of S comes from the fp32 normalized row dotted with itself
    red[t] = r * r;
    __syncthreads();
    #pragma unroll
    for (int s = 128; s > 0; s >>= 1) {
        if (t < s) red[t] += red[t + s];
        __syncthreads();
    }
    if (t == 0) selfdot[row] = red[0];
}

// --------------------------- K2: positive-pair dots ------------------------
__global__ __launch_bounds__(256)
void ntx_posdot_kernel(const __bf16* __restrict__ Zh,
                       const __bf16* __restrict__ Zl,
                       float* __restrict__ posdot) {
    const int p  = blockIdx.x;           // pair index, rows 2p / 2p+1
    const int t  = threadIdx.x;
    const int i0 = 2 * p, i1 = 2 * p + 1;
    const float a = (float)Zh[i0 * DIM + t] + (float)Zl[i0 * DIM + t];
    const float b = (float)Zh[i1 * DIM + t] + (float)Zl[i1 * DIM + t];

    __shared__ float red[256];
    red[t] = a * b;
    __syncthreads();
    #pragma unroll
    for (int s = 128; s > 0; s >>= 1) {
        if (t < s) red[t] += red[t + s];
        __syncthreads();
    }
    if (t == 0) { posdot[i0] = red[0]; posdot[i1] = red[0]; }
}

// ----------------- K3: WMMA GEMM + fused exp/rowsum epilogue ---------------
// Wave -> 16-row strip. Workgroup (8 waves) -> 128 rows. blockIdx.y splits the
// 8192 columns into CSPLIT deterministic partial sums. All data L2-resident.
__global__ __launch_bounds__(256)
void ntx_sim_kernel(const __bf16* __restrict__ Zh,
                    const __bf16* __restrict__ Zl,
                    float* __restrict__ parts) {
    const int lane    = threadIdx.x & 31;
    const int wave    = threadIdx.x >> 5;
    const int rowBase = blockIdx.x * 128 + wave * 16;
    const int colBase = blockIdx.y * (NROWS / CSPLIT);

    // A fragment addressing (16-bit A 16x32, ISA 7.12.2):
    // low lanes: K {c..c+7, c+16..c+23} with c=0; high lanes: c=8
    const int m     = rowBase + (lane & 15);
    const int abase = (lane >= 16) ? 8  : 0;
    // B fragment: lane holds column n, K {c..c+7, c+16..c+23}, c=0 low / 8 high
    const int bbase = (lane >= 16) ? 8  : 0;

    // Preload the full K=256 A panel (hi+lo) into registers: reused 64x.
    Frag Ah[8], Al[8];
    const __bf16* rowH = Zh + m * DIM;
    const __bf16* rowL = Zl + m * DIM;
    #pragma unroll
    for (int ks = 0; ks < 8; ++ks) {
        const __bf16* ph = rowH + ks * 32 + abase;
        const __bf16* pl = rowL + ks * 32 + abase;
        Ah[ks].q[0] = *(const v4u*)(ph);
        Ah[ks].q[1] = *(const v4u*)(ph + 16);
        Al[ks].q[0] = *(const v4u*)(pl);
        Al[ks].q[1] = *(const v4u*)(pl + 16);
    }

    float partial[8];
    #pragma unroll
    for (int r = 0; r < 8; ++r) partial[r] = 0.0f;

    for (int ct = 0; ct < (NROWS / CSPLIT) / 16; ++ct) {
        const int n = colBase + ct * 16 + (lane & 15);
        const __bf16* colH = Zh + n * DIM + bbase;
        const __bf16* colL = Zl + n * DIM + bbase;

        v8f acc = {};
        #pragma unroll
        for (int ks = 0; ks < 8; ++ks) {
            Frag Bh, Bl;
            Bh.q[0] = *(const v4u*)(colH + ks * 32);
            Bh.q[1] = *(const v4u*)(colH + ks * 32 + 16);
            Bl.q[0] = *(const v4u*)(colL + ks * 32);
            Bl.q[1] = *(const v4u*)(colL + ks * 32 + 16);
            // Z Z^T ~= Zh Zh^T + Zh Zl^T + Zl Zh^T   (drop lo*lo term)
            acc = __builtin_amdgcn_wmma_f32_16x16x32_bf16(
                      false, Ah[ks].bf, false, Bh.bf, (short)0, acc, false, false);
            acc = __builtin_amdgcn_wmma_f32_16x16x32_bf16(
                      false, Ah[ks].bf, false, Bl.bf, (short)0, acc, false, false);
            acc = __builtin_amdgcn_wmma_f32_16x16x32_bf16(
                      false, Al[ks].bf, false, Bh.bf, (short)0, acc, false, false);
        }

        // Fused epilogue: S = exp(dot / 0.5); accumulate per-lane row partials.
        // C/D layout: lane l, VGPR r -> C[m][n], m = r + 8*(l>=16), n = l&15.
        #pragma unroll
        for (int r = 0; r < 8; ++r)
            partial[r] += __expf(2.0f * acc[r]);
    }

    // Reduce over the 16 lanes of each half-wave (the N dimension).
    #pragma unroll
    for (int r = 0; r < 8; ++r) {
        float v = partial[r];
        v += __shfl_xor(v, 1, 16);
        v += __shfl_xor(v, 2, 16);
        v += __shfl_xor(v, 4, 16);
        v += __shfl_xor(v, 8, 16);
        if ((lane & 15) == 0) {
            const int row = rowBase + r + ((lane >> 4) << 3);
            parts[row * CSPLIT + blockIdx.y] = v;   // exclusive write: no atomics
        }
    }
}

// --------------------------- K4: final loss --------------------------------
__global__ __launch_bounds__(256)
void ntx_loss_kernel(const float* __restrict__ parts,
                     const float* __restrict__ selfdot,
                     const float* __restrict__ posdot,
                     float* __restrict__ out) {
    const int t = threadIdx.x;
    float acc = 0.0f;
    for (int i = t; i < NROWS; i += 256) {
        float rs = 0.0f;
        #pragma unroll
        for (int c = 0; c < CSPLIT; ++c) rs += parts[i * CSPLIT + c];
        const float diag = __expf(2.0f * selfdot[i]);
        acc += -2.0f * posdot[i] + logf(rs - diag + 1e-8f);
    }
    __shared__ float red[256];
    red[t] = acc;
    __syncthreads();
    #pragma unroll
    for (int s = 128; s > 0; s >>= 1) {
        if (t < s) red[t] += red[t + s];
        __syncthreads();
    }
    if (t == 0) out[0] = red[0] / (float)NROWS;
}

// --------------------------- host launcher ---------------------------------
extern "C" void kernel_launch(void* const* d_in, const int* in_sizes, int n_in,
                              void* d_out, int out_size, void* d_ws, size_t ws_size,
                              hipStream_t stream) {
    const float* z1 = (const float*)d_in[0];
    const float* z2 = (const float*)d_in[1];
    float* out = (float*)d_out;

    // Workspace layout (~8.4 MB total):
    char* ws = (char*)d_ws;
    __bf16* Zh      = (__bf16*)ws;                         // 8192*256 bf16 = 4 MB
    __bf16* Zl      = Zh + (size_t)NROWS * DIM;            // 4 MB
    float*  selfdot = (float*)(Zl + (size_t)NROWS * DIM);  // 32 KB
    float*  posdot  = selfdot + NROWS;                     // 32 KB
    float*  parts   = posdot + NROWS;                      // 8192*8 f32 = 256 KB

    ntx_normalize_kernel<<<NROWS, 256, 0, stream>>>(z1, z2, Zh, Zl, selfdot);
    ntx_posdot_kernel<<<NROWS / 2, 256, 0, stream>>>(Zh, Zl, posdot);
    dim3 grid(NROWS / 128, CSPLIT);
    ntx_sim_kernel<<<grid, 256, 0, stream>>>(Zh, Zl, parts);
    ntx_loss_kernel<<<1, 256, 0, stream>>>(parts, selfdot, posdot, out);
}
